// AdviceTransformer_67577015435504
// MI455X (gfx1250) — matrix operand order repaired and checked by reading, not compiled, not run on hardware
//
#include <hip/hip_runtime.h>
#include <hip/hip_bf16.h>

// ---------------------------------------------------------------------------
// GPT forward pass for MI455X (gfx1250, wave32, WMMA).
// bf16 end-to-end GEMM pipeline: LayerNorm/attention emit bf16 activations,
// per-layer kernels convert weights f32->bf16 (L2-resident, ~free). The WMMA
// GEMM double-buffers LDS tiles: while 8 v_wmma_f32_16x16x32_bf16 consume
// buffer p, GLOBAL_LOAD_ASYNC_TO_LDS_B128 (ASYNCcnt) streams the next A-tile
// and two-phase b128 loads + transpose scatter stage the next B-tile into
// buffer 1-p. Epilogue variants are template parameters (straight-line).
// ---------------------------------------------------------------------------

#define TD   1024      // T (sequence length)
#define DM   1024      // D (model dim)
#define HH   16        // heads
#define HD   64        // head dim
#define LL   8         // layers
#define VV   50257     // vocab
#define BB   4         // batch
#define MM   4096      // B*T rows
#define ATT_SCALE 0.07216878364870323f   // 1/sqrt(3*D/H) = 1/sqrt(192)

typedef __attribute__((ext_vector_type(8)))  float  v8f;
typedef __attribute__((ext_vector_type(16))) __bf16 v16bf;
typedef __attribute__((ext_vector_type(8)))  __bf16 v8bf;
typedef __attribute__((ext_vector_type(4)))  __bf16 v4bf;
typedef __attribute__((ext_vector_type(2)))  __bf16 v2bf;
// reduced-alignment vectors for rows whose stride is not 16B aligned (N=50257)
typedef float  float4a __attribute__((ext_vector_type(4), aligned(4)));
typedef __bf16 v8bfa   __attribute__((ext_vector_type(8), aligned(4)));

union FragBF { v16bf v; v8bf h[2]; };

__device__ __forceinline__ __bf16 f2bf(float f) {
  unsigned u = __builtin_bit_cast(unsigned, f);
  unsigned r = u + 0x7FFFu + ((u >> 16) & 1u);   // round-to-nearest-even
  return __builtin_bit_cast(__bf16, (unsigned short)(r >> 16));
}

__device__ __forceinline__ int imin(int a, int b) { return a < b ? a : b; }

// --------------------------- f32 -> bf16 convert ---------------------------
__global__ __launch_bounds__(256) void convert_bf16_kernel(
    const float* __restrict__ in, __bf16* __restrict__ out, int n4) {
  const int i = blockIdx.x * 256 + threadIdx.x;
  if (i >= n4) return;
  const float4 v = ((const float4*)in)[i];
  v4bf b; b[0] = f2bf(v.x); b[1] = f2bf(v.y); b[2] = f2bf(v.z); b[3] = f2bf(v.w);
  ((v4bf*)out)[i] = b;
}

// --------------------------- embedding -------------------------------------
__global__ __launch_bounds__(256) void embed_kernel(
    const int* __restrict__ idx, const float* __restrict__ tok,
    const float* __restrict__ pos, float* __restrict__ x) {
  const int row = blockIdx.x;            // 0..4095
  const int t   = row & (TD - 1);
  const int token = idx[row];
  const float4 a = ((const float4*)(tok + (size_t)token * DM))[threadIdx.x];
  const float4 b = ((const float4*)(pos + (size_t)t     * DM))[threadIdx.x];
  float4 o; o.x = a.x + b.x; o.y = a.y + b.y; o.z = a.z + b.z; o.w = a.w + b.w;
  ((float4*)(x + (size_t)row * DM))[threadIdx.x] = o;
}

// --------------------------- layernorm (bf16 out) --------------------------
__global__ __launch_bounds__(256) void layernorm_kernel(
    const float* __restrict__ x, const float* __restrict__ g,
    const float* __restrict__ b, __bf16* __restrict__ out) {
  __shared__ float s1[256], s2[256];
  const int row = blockIdx.x;
  const float4 v = ((const float4*)(x + (size_t)row * DM))[threadIdx.x];
  float s = v.x + v.y + v.z + v.w;
  float q = v.x * v.x + v.y * v.y + v.z * v.z + v.w * v.w;
  s1[threadIdx.x] = s; s2[threadIdx.x] = q;
  __syncthreads();
  for (int st = 128; st > 0; st >>= 1) {
    if (threadIdx.x < st) {
      s1[threadIdx.x] += s1[threadIdx.x + st];
      s2[threadIdx.x] += s2[threadIdx.x + st];
    }
    __syncthreads();
  }
  const float mu  = s1[0] * (1.0f / DM);
  const float var = s2[0] * (1.0f / DM) - mu * mu;
  const float inv = rsqrtf(var + 1e-5f);
  const float4 gg = ((const float4*)g)[threadIdx.x];
  const float4 bb = ((const float4*)b)[threadIdx.x];
  v4bf o;
  o[0] = f2bf((v.x - mu) * inv * gg.x + bb.x);
  o[1] = f2bf((v.y - mu) * inv * gg.y + bb.y);
  o[2] = f2bf((v.z - mu) * inv * gg.z + bb.z);
  o[3] = f2bf((v.w - mu) * inv * gg.w + bb.w);
  ((v4bf*)(out + (size_t)row * DM))[threadIdx.x] = o;
}

// --------------------------- WMMA GEMM -------------------------------------
// Out[M,N] = A_bf16[M,K] @ B[K,N] (+bias) (+residual) (relu?).
// Requires M % 64 == 0 and K % 32 == 0 (always true here).
// Double-buffered LDS: async A copy + two-phase B scatter for tile kt+1
// overlap the 8 WMMAs on tile kt. 128 threads = 4 waves; wave tile 32x64.
template <bool BIAS, bool RES, bool RELU, bool NT, bool OUT_BF16, bool B_BF16,
          bool NTAIL>
__global__ __launch_bounds__(128) void gemm_wmma(
    const __bf16* __restrict__ A, const void* __restrict__ Bv,
    const float* __restrict__ bias, const float* __restrict__ residual,
    void* __restrict__ Out, int Mn, int Nn, int Kn) {
  __shared__ __bf16 As[2][64][40];    // 64 rows x 32 k (80B stride, 16B mult)
  __shared__ __bf16 Bs[2][128][40];   // 128 cols x 32 k (transposed)

  const int tid   = threadIdx.x;
  const int lane  = tid & 31;
  const int wave  = tid >> 5;
  const int waveM = (wave >> 1) * 32;   // 0 / 32
  const int waveN = (wave & 1) * 64;    // 0 / 64
  const int mBase = blockIdx.y * 64;
  const int nBase = blockIdx.x * 128;

  // ---- tile staging helpers ----
  auto stageA = [&](int buf, int k0) {
    // 64x32 bf16, GLOBAL_LOAD_ASYNC_TO_LDS_B128, 16B per lane
    #pragma unroll
    for (int i = 0; i < 2; ++i) {
      const int e = tid + i * 128;          // 0..255 16B-chunks
      const int r = e >> 2, c = (e & 3) * 8;
      const unsigned ldsoff =
          (unsigned)(unsigned long long)&As[buf][r][c];
      const unsigned long long ga =
          (unsigned long long)&A[(size_t)(mBase + r) * Kn + k0 + c];
      asm volatile("global_load_async_to_lds_b128 %0, %1, off"
                   :: "v"(ldsoff), "v"(ga) : "memory");
    }
  };
  auto stageB = [&](int buf, int k0) {
    // 32x128, two-phase: batch all global loads, then transpose scatter.
    if (B_BF16) {
      const __bf16* Bb = (const __bf16*)Bv;
      v8bfa tmp[4];
      #pragma unroll
      for (int i = 0; i < 4; ++i) {
        const int e = tid + i * 128;        // 0..511 8-elem chunks
        const int r = e >> 4;               // k row 0..31
        const int c8 = e & 15;              // n group
        const int n = NTAIL ? imin(nBase + c8 * 8, Nn - 8) : nBase + c8 * 8;
        tmp[i] = *(const v8bfa*)&Bb[(size_t)(k0 + r) * Nn + n];
      }
      #pragma unroll
      for (int i = 0; i < 4; ++i) {
        const int e = tid + i * 128;
        const int r = e >> 4;
        const int c8 = e & 15;
        #pragma unroll
        for (int j = 0; j < 8; ++j) Bs[buf][c8 * 8 + j][r] = tmp[i][j];
      }
    } else {
      const float* Bf = (const float*)Bv;
      float4a tmp[8];
      #pragma unroll
      for (int i = 0; i < 8; ++i) {
        const int e = tid + i * 128;        // 0..1023 float4s
        const int r = e >> 5;               // k row 0..31
        const int c4 = e & 31;              // n group
        const int n = NTAIL ? imin(nBase + c4 * 4, Nn - 4) : nBase + c4 * 4;
        tmp[i] = *(const float4a*)&Bf[(size_t)(k0 + r) * Nn + n];
      }
      #pragma unroll
      for (int i = 0; i < 8; ++i) {
        const int e = tid + i * 128;
        const int r = e >> 5;
        const int c4 = e & 31;
        Bs[buf][c4 * 4 + 0][r] = f2bf(tmp[i].x);
        Bs[buf][c4 * 4 + 1][r] = f2bf(tmp[i].y);
        Bs[buf][c4 * 4 + 2][r] = f2bf(tmp[i].z);
        Bs[buf][c4 * 4 + 3][r] = f2bf(tmp[i].w);
      }
    }
  };

  v8f acc[2][4] = {};
  const int nK = Kn >> 5;

  // prologue: stage tile 0 into buffer 0
  stageA(0, 0);
  stageB(0, 0);
  asm volatile("s_wait_asynccnt 0x0" ::: "memory");
  __syncthreads();

  for (int kt = 0; kt < nK; ++kt) {
    const int cur = kt & 1;

    // stage next tile into the other buffer (overlaps with WMMAs below)
    if (kt + 1 < nK) {
      stageA(cur ^ 1, (kt + 1) << 5);
      stageB(cur ^ 1, (kt + 1) << 5);
    }

    // Fragment loads per ISA 7.12.2 layouts.
    // A (16x32 bf16): lane&15 = row, lane>>4 selects K half (0/8 and 16/24).
    // B (transposed): lane&15 = col, lane>>4 selects K half (0/8 and 16/24).
    FragBF a[2], b[4];
    const int kb_a = (lane >> 4) * 8;
    const int kb_b = (lane >> 4) * 8;
    const int ml   = lane & 15;
    #pragma unroll
    for (int mi = 0; mi < 2; ++mi) {
      const int rrow = waveM + mi * 16 + ml;
      a[mi].h[0] = *(const v8bf*)&As[cur][rrow][kb_a];
      a[mi].h[1] = *(const v8bf*)&As[cur][rrow][16 + kb_a];
    }
    #pragma unroll
    for (int ni = 0; ni < 4; ++ni) {
      const int ncol = waveN + ni * 16 + ml;
      b[ni].h[0] = *(const v8bf*)&Bs[cur][ncol][kb_b];
      b[ni].h[1] = *(const v8bf*)&Bs[cur][ncol][16 + kb_b];
    }
    #pragma unroll
    for (int mi = 0; mi < 2; ++mi)
      #pragma unroll
      for (int ni = 0; ni < 4; ++ni)
        acc[mi][ni] = __builtin_amdgcn_wmma_f32_16x16x32_bf16(
            false, a[mi].v, false, b[ni].v, (short)0, acc[mi][ni],
            false, false);

    // retire pipeline stage: async A for kt+1 done, all waves synced
    asm volatile("s_wait_asynccnt 0x0" ::: "memory");
    __syncthreads();
  }

  // epilogue: C/D layout -> lane&15 = N, lane>>4 picks M group (0-7 / 8-15).
  // M is grid-exact; only the LM head (NTAIL) needs the N guard.
  const int group = lane >> 4;
  const int nl    = lane & 15;
  #pragma unroll
  for (int mi = 0; mi < 2; ++mi) {
    #pragma unroll
    for (int ni = 0; ni < 4; ++ni) {
      const int n = nBase + waveN + ni * 16 + nl;
      if (NTAIL && n >= Nn) continue;
      const float bval = BIAS ? bias[n] : 0.0f;
      #pragma unroll
      for (int vv = 0; vv < 8; ++vv) {
        const int m = mBase + waveM + mi * 16 + group * 8 + vv;
        float val = acc[mi][ni][vv] + bval;
        if (RES) val += residual[(size_t)m * Nn + n];
        if (RELU) val = fmaxf(val, 0.0f);
        if (OUT_BF16) {
          ((__bf16*)Out)[(size_t)m * Nn + n] = f2bf(val);
        } else if (NT) {
          __builtin_nontemporal_store(val, &((float*)Out)[(size_t)m * Nn + n]);
        } else {
          ((float*)Out)[(size_t)m * Nn + n] = val;
        }
      }
    }
  }
}

// --------------------------- attention (bf16 out) --------------------------
// One wave per query row; online softmax over keys 0..t; each lane owns 2 of
// the 64 head dims. qkv layout: row*3072 + {0,1024,2048} + h*64.
__global__ __launch_bounds__(256) void attention_kernel(
    const float* __restrict__ qkv, __bf16* __restrict__ y) {
  const int wave = threadIdx.x >> 5;
  const int lane = threadIdx.x & 31;
  const int t = blockIdx.x * 8 + wave;
  const int h = blockIdx.y;
  const int b = blockIdx.z;
  const int row = b * TD + t;

  const float* qp = qkv + (size_t)row * (3 * DM) + h * HD;
  const float q0 = qp[lane * 2];
  const float q1 = qp[lane * 2 + 1];

  float m = -INFINITY, l = 0.0f, o0 = 0.0f, o1 = 0.0f;
  for (int s = 0; s <= t; ++s) {
    const float* kp = qkv + (size_t)(b * TD + s) * (3 * DM) + DM + h * HD;
    float p = q0 * kp[lane * 2] + q1 * kp[lane * 2 + 1];
    #pragma unroll
    for (int off = 16; off > 0; off >>= 1) p += __shfl_xor(p, off, 32);
    p *= ATT_SCALE;
    const float mNew = fmaxf(m, p);
    const float corr = __expf(m - mNew);   // 0 on first iter (m = -inf)
    const float w    = __expf(p - mNew);
    l = l * corr + w;
    const float* vp = qkv + (size_t)(b * TD + s) * (3 * DM) + 2 * DM + h * HD;
    o0 = o0 * corr + w * vp[lane * 2];
    o1 = o1 * corr + w * vp[lane * 2 + 1];
    m = mNew;
  }
  const float inv = 1.0f / l;
  v2bf o; o[0] = f2bf(o0 * inv); o[1] = f2bf(o1 * inv);
  *(v2bf*)&y[(size_t)row * DM + h * HD + lane * 2] = o;
}

// --------------------------- loss ------------------------------------------
__global__ __launch_bounds__(256) void loss_kernel(
    const float* __restrict__ logits, const int* __restrict__ targets,
    float* __restrict__ accum, int Vn) {
  __shared__ float red[256];
  const int row = blockIdx.x;
  const float* lg = logits + (size_t)row * Vn;

  float mx = -INFINITY;
  for (int i = threadIdx.x; i < Vn; i += 256) mx = fmaxf(mx, lg[i]);
  red[threadIdx.x] = mx; __syncthreads();
  for (int st = 128; st > 0; st >>= 1) {
    if (threadIdx.x < st)
      red[threadIdx.x] = fmaxf(red[threadIdx.x], red[threadIdx.x + st]);
    __syncthreads();
  }
  mx = red[0]; __syncthreads();

  float se = 0.0f;
  for (int i = threadIdx.x; i < Vn; i += 256) se += __expf(lg[i] - mx);
  red[threadIdx.x] = se; __syncthreads();
  for (int st = 128; st > 0; st >>= 1) {
    if (threadIdx.x < st) red[threadIdx.x] += red[threadIdx.x + st];
    __syncthreads();
  }

  if (threadIdx.x == 0) {
    const int tgt = targets[row];
    const float nll = -(lg[tgt] - mx - __logf(red[0]));
    if (tgt != 0) {            // PAD == 0
      atomicAdd(&accum[0], nll);
      atomicAdd(&accum[1], 1.0f);
    }
  }
}

__global__ void init_kernel(float* accum) {
  if (threadIdx.x < 2) accum[threadIdx.x] = 0.0f;
}

__global__ void finalize_kernel(const float* accum, float* lossOut) {
  lossOut[0] = accum[0] / fmaxf(accum[1], 1.0f);
}

// --------------------------- host ------------------------------------------
extern "C" void kernel_launch(void* const* d_in, const int* in_sizes, int n_in,
                              void* d_out, int out_size, void* d_ws,
                              size_t ws_size, hipStream_t stream) {
  const int*   idx     = (const int*)d_in[0];
  const int*   targets = (const int*)d_in[1];
  const float* tok_emb = (const float*)d_in[2];
  const float* pos_emb = (const float*)d_in[3];
  const float* ln1_g   = (const float*)d_in[4];
  const float* ln1_b   = (const float*)d_in[5];
  const float* wqkv    = (const float*)d_in[6];
  const float* bqkv    = (const float*)d_in[7];
  const float* wproj   = (const float*)d_in[8];
  const float* bproj   = (const float*)d_in[9];
  const float* ln2_g   = (const float*)d_in[10];
  const float* ln2_b   = (const float*)d_in[11];
  const float* w1      = (const float*)d_in[12];
  const float* b1      = (const float*)d_in[13];
  const float* w2      = (const float*)d_in[14];
  const float* b2      = (const float*)d_in[15];
  const float* lm_w    = (const float*)d_in[16];

  // workspace layout: ~128 MB
  float*  ws    = (float*)d_ws;
  float*  accum = ws;                                    // 16 floats
  float*  x     = ws + 16;                               // M*D   f32
  float*  qkv   = x + (size_t)MM * DM;                   // M*3D  f32
  __bf16* hb    = (__bf16*)(qkv + (size_t)MM * 3 * DM);  // M*D   bf16
  __bf16* yb    = hb  + (size_t)MM * DM;                 // M*D   bf16
  __bf16* m1b   = yb  + (size_t)MM * DM;                 // M*4D  bf16
  __bf16* xb    = m1b + (size_t)MM * 4 * DM;             // M*D   bf16
  __bf16* wbuf  = xb  + (size_t)MM * DM;                 // 4M    bf16 (weights)

  float* logits  = (float*)d_out;
  float* lossOut = logits + (out_size - 1);

  init_kernel<<<1, 32, 0, stream>>>(accum);
  embed_kernel<<<MM, 256, 0, stream>>>(idx, tok_emb, pos_emb, x);

  const dim3 gemmBlk(128);
  for (int l = 0; l < LL; ++l) {
    // attn: ln1 -> qkv gemm -> attention -> proj gemm (+residual)
    layernorm_kernel<<<MM, 256, 0, stream>>>(x, ln1_g + l * DM, ln1_b + l * DM, hb);
    convert_bf16_kernel<<<(DM * 3 * DM / 4) / 256, 256, 0, stream>>>(
        wqkv + (size_t)l * DM * 3 * DM, wbuf, DM * 3 * DM / 4);
    gemm_wmma<true, false, false, false, false, true, false>
        <<<dim3(3 * DM / 128, MM / 64), gemmBlk, 0, stream>>>(
        hb, wbuf, bqkv + (size_t)l * 3 * DM, nullptr, qkv, MM, 3 * DM, DM);
    attention_kernel<<<dim3(TD / 8, HH, BB), 256, 0, stream>>>(qkv, yb);
    convert_bf16_kernel<<<(DM * DM / 4) / 256, 256, 0, stream>>>(
        wproj + (size_t)l * DM * DM, wbuf, DM * DM / 4);
    gemm_wmma<true, true, false, false, false, true, false>
        <<<dim3(DM / 128, MM / 64), gemmBlk, 0, stream>>>(
        yb, wbuf, bproj + (size_t)l * DM, x, x, MM, DM, DM);
    // mlp: ln2 -> w1 gemm (relu, bf16 out) -> w2 gemm (+residual)
    layernorm_kernel<<<MM, 256, 0, stream>>>(x, ln2_g + l * DM, ln2_b + l * DM, hb);
    convert_bf16_kernel<<<(DM * 4 * DM / 4) / 256, 256, 0, stream>>>(
        w1 + (size_t)l * DM * 4 * DM, wbuf, DM * 4 * DM / 4);
    gemm_wmma<true, false, true, false, true, true, false>
        <<<dim3(4 * DM / 128, MM / 64), gemmBlk, 0, stream>>>(
        hb, wbuf, b1 + (size_t)l * 4 * DM, nullptr, m1b, MM, 4 * DM, DM);
    convert_bf16_kernel<<<(DM * 4 * DM / 4) / 256, 256, 0, stream>>>(
        w2 + (size_t)l * 4 * DM * DM, wbuf, DM * 4 * DM / 4);
    gemm_wmma<true, true, false, false, false, true, false>
        <<<dim3(DM / 128, MM / 64), gemmBlk, 0, stream>>>(
        m1b, wbuf, b2 + (size_t)l * DM, x, x, MM, DM, 4 * DM);
  }

  // LM head: bf16(x) @ f32 lm_w (converted in-kernel) -> logits, NT stores
  convert_bf16_kernel<<<(MM * DM / 4) / 256, 256, 0, stream>>>(x, xb, MM * DM / 4);
  gemm_wmma<false, false, false, true, false, false, true>
      <<<dim3((VV + 127) / 128, MM / 64), gemmBlk, 0, stream>>>(
      xb, lm_w, nullptr, nullptr, logits, MM, VV, DM);

  loss_kernel<<<MM, 256, 0, stream>>>(logits, targets, accum, VV);
  finalize_kernel<<<1, 1, 0, stream>>>(accum, lossOut);
}
